// Model_76209899700973
// MI455X (gfx1250) — matrix operand order repaired and checked by reading, not compiled, not run on hardware
//
#include <hip/hip_runtime.h>

typedef __attribute__((ext_vector_type(16))) _Float16 v16h;
typedef __attribute__((ext_vector_type(8)))  _Float16 v8h;
typedef __attribute__((ext_vector_type(8)))  float    v8f;

#define NWAVES 8
#define NTHREADS 256

// ---------------- layer tables ----------------
// Weights prepacked in WMMA *fragment order*:
//   wpk[ ((tn*Ksteps + ks)*32 + lane)*16 + e ]
// per CDNA5 B-matrix (32x16 f16) layout:
//   n = tn*16 + (lane&15), k = ks*32 + (lane<16 ? 0 : 8) + (e<8 ? e : e+8)
constexpr int WOFF[8] = {0, 1024, 10240, 19456, 28672, 47104, 83968, 968704};
constexpr int WTOT    = 972800;                 // halves
constexpr int BOFF[8] = {0, 32, 64, 96, 128, 192, 256, 384};
constexpr size_t BIAS_BYTE = (size_t)WTOT * 2;

// ---------------- prepack: fold BN, emit fragment-ordered f16 weights ----------------
__global__ void prepack_layer(const float* __restrict__ w, const float* __restrict__ b,
                              const float* __restrict__ gamma, const float* __restrict__ beta,
                              const float* __restrict__ mean, const float* __restrict__ var,
                              int Cin, int KH, int KW, int Cout, int CoutP,
                              _Float16* __restrict__ wpk, float* __restrict__ bpk, int has_bn) {
    const int K      = Cin * KH * KW;
    const int Ksteps = (K + 31) / 32;
    const int Nt     = CoutP / 16;
    const int total  = Nt * Ksteps * 512;
    const int idx    = blockIdx.x * blockDim.x + threadIdx.x;
    if (idx < total) {
        const int e    = idx & 15;
        const int lane = (idx >> 4) & 31;
        const int ks   = (idx >> 9) % Ksteps;
        const int tn   = idx / (Ksteps << 9);
        const int n    = tn * 16 + (lane & 15);
        const int k    = ks * 32 + ((lane < 16) ? 0 : 8) + ((e < 8) ? e : e + 8);
        float v = 0.0f;
        if (k < K && n < Cout) {
            const int c  = k % Cin;
            const int kw = (k / Cin) % KW;
            const int kh = k / (Cin * KW);
            const float s = has_bn ? gamma[n] * rsqrtf(var[n] + 1e-5f) : 1.0f;
            v = w[((size_t)(n * Cin + c) * KH + kh) * KW + kw] * s;
        }
        wpk[idx] = (_Float16)v;
    }
    if (idx < CoutP) {
        float bb = 0.0f;
        if (idx < Cout) {
            const float s = has_bn ? gamma[idx] * rsqrtf(var[idx] + 1e-5f) : 1.0f;
            bb = has_bn ? (b[idx] - mean[idx]) * s + beta[idx] : b[idx];
        }
        bpk[idx] = bb;
    }
}

// ---------------- templated implicit-GEMM conv layer (LDS act -> LDS act) ----------------
// Each wave owns one M-tile x two N-tiles: A fragment loaded once, used by 2 WMMAs.
template <int InH, int InW, int Cin, int OutH, int OutW, int CoutP,
          int KH, int KW, int PAD, bool RELU>
__device__ __forceinline__ void conv_wmma(const _Float16* __restrict__ in,
                                          _Float16* __restrict__ out,
                                          const _Float16* __restrict__ wpk,
                                          const float* __restrict__ bpk) {
    constexpr int M      = OutH * OutW;
    constexpr int K      = Cin * KH * KW;
    constexpr int Ksteps = (K + 31) / 32;
    constexpr int Mt     = (M + 15) / 16;
    constexpr int Nt     = CoutP / 16;      // always even here
    constexpr int NG     = Nt / 2;          // N-tile pair groups
    constexpr int KWC    = KW * Cin;

    const int lane   = threadIdx.x & 31;
    const int half16 = lane & 15;
    const bool hi    = lane >= 16;

    for (int grp = (int)(threadIdx.x >> 5); grp < Mt * NG; grp += NWAVES) {
        const int tm   = grp / NG;
        const int tn0  = (grp % NG) * 2;            // first of the N-tile pair
        const int mrow = tm * 16 + half16;          // A-matrix row this lane feeds
        const bool mv  = (mrow < M);
        const int oh   = mrow / OutW;
        const int ow   = mrow % OutW;
        const int lanebase = (oh * InW + ow) * Cin + (hi ? 8 : 0);

        v8f acc0 = {0.f, 0.f, 0.f, 0.f, 0.f, 0.f, 0.f, 0.f};
        v8f acc1 = {0.f, 0.f, 0.f, 0.f, 0.f, 0.f, 0.f, 0.f};
        const v8h* __restrict__ bfrag0 =
            reinterpret_cast<const v8h*>(wpk + ((size_t)(tn0    ) * Ksteps * 32 + lane) * 16);
        const v8h* __restrict__ bfrag1 =
            reinterpret_cast<const v8h*>(wpk + ((size_t)(tn0 + 1) * Ksteps * 32 + lane) * 16);

        for (int ks = 0; ks < Ksteps; ++ks) {
            // ---- A fragment (shared by both N-tiles)
            v16h a;
            if constexpr (Cin % 32 == 0) {
                // 32-aligned K-run sits inside one (kh,kw) tap, channels contiguous:
                // per lane two contiguous 16B-aligned v8h runs -> 2x ds_load_b128
                const int khkw = (ks * 32) / Cin;        // wave-uniform
                const int c0   = (ks * 32) % Cin;
                const int kh   = khkw / KW;
                const int kw   = khkw % KW;
                const int off  = (kh * InW + kw - PAD * (InW + 1)) * Cin + c0;
                bool valid = mv;
                if constexpr (PAD > 0) {
                    const int ih = oh - PAD + kh;
                    const int iw = ow - PAD + kw;
                    valid = valid && ((unsigned)ih < (unsigned)InH) &&
                                     ((unsigned)iw < (unsigned)InW);
                }
                v8h a0 = (v8h)(_Float16)0.0f;
                v8h a1 = (v8h)(_Float16)0.0f;
                if (valid) {
                    const v8h* p = reinterpret_cast<const v8h*>(in + lanebase + off);
                    a0 = p[0];      // K = run + {0..7}
                    a1 = p[2];      // K = run + {16..23}
                }
                a = __builtin_shufflevector(a0, a1,
                    0, 1, 2, 3, 4, 5, 6, 7, 8, 9, 10, 11, 12, 13, 14, 15);
            } else {
                // generic gather (layer 1 only: Cin=1, K=9, Ksteps=1) - branch-free
#pragma unroll
                for (int v = 0; v < 16; ++v) {
                    const int k = ks * 32 + (hi ? 8 : 0) +
                                  ((v < 8) ? 0 : 16) + (v & 7);
                    _Float16 av = (_Float16)0.0f;
                    if (k < K) {     // compile-time after unroll (Ksteps==1)
                        const int kh = k / KWC;
                        const int r  = k - kh * KWC;
                        const int kw = r / Cin;
                        const int c  = r - kw * Cin;
                        const int ih = oh - PAD + kh;
                        const int iw = ow - PAD + kw;
                        const bool ok = mv && ((unsigned)ih < (unsigned)InH) &&
                                              ((unsigned)iw < (unsigned)InW);
                        const int addr = ok ? (ih * InW + iw) * Cin + c : 0;
                        const _Float16 ld = in[addr];
                        av = ok ? ld : (_Float16)0.0f;
                    }
                    a[v] = av;
                }
            }

            // ---- B fragments: contiguous global b128 pairs (L2-resident weights)
            {
                const v8h b0 = bfrag0[ks * 64 + 0];
                const v8h b1 = bfrag0[ks * 64 + 1];
                const v16h b = __builtin_shufflevector(b0, b1,
                    0, 1, 2, 3, 4, 5, 6, 7, 8, 9, 10, 11, 12, 13, 14, 15);
                acc0 = __builtin_amdgcn_wmma_f32_16x16x32_f16(
                    false, a, false, b, (short)0, acc0, false, false);
            }
            {
                const v8h b0 = bfrag1[ks * 64 + 0];
                const v8h b1 = bfrag1[ks * 64 + 1];
                const v16h b = __builtin_shufflevector(b0, b1,
                    0, 1, 2, 3, 4, 5, 6, 7, 8, 9, 10, 11, 12, 13, 14, 15);
                acc1 = __builtin_amdgcn_wmma_f32_16x16x32_f16(
                    false, a, false, b, (short)0, acc1, false, false);
            }
        }

        // ---- epilogue: folded bias (+ReLU), f32 -> f16, NHWC store to LDS
        const int n0   = tn0 * 16 + half16;
        const float bnA = bpk[n0];
        const float bnB = bpk[n0 + 16];
        const int mb   = tm * 16 + (hi ? 8 : 0);
        if ((M % 16 == 0) || (tm != Mt - 1)) {
#pragma unroll
            for (int v = 0; v < 8; ++v) {
                float vA = acc0[v] + bnA;
                float vB = acc1[v] + bnB;
                if (RELU) { vA = fmaxf(vA, 0.0f); vB = fmaxf(vB, 0.0f); }
                _Float16* row = out + (size_t)(mb + v) * CoutP + n0;
                row[0]  = (_Float16)vA;
                row[16] = (_Float16)vB;
            }
        } else {
#pragma unroll
            for (int v = 0; v < 8; ++v) {
                const int m = mb + v;
                if (m < M) {
                    float vA = acc0[v] + bnA;
                    float vB = acc1[v] + bnB;
                    if (RELU) { vA = fmaxf(vA, 0.0f); vB = fmaxf(vB, 0.0f); }
                    _Float16* row = out + (size_t)m * CoutP + n0;
                    row[0]  = (_Float16)vA;
                    row[16] = (_Float16)vB;
                }
            }
        }
    }
}

// ---------------- 2x1 maxpool over H (NHWC in LDS) ----------------
template <int InH, int InW, int C>
__device__ __forceinline__ void maxpool2(const _Float16* __restrict__ in,
                                         _Float16* __restrict__ out) {
    constexpr int OutH = InH / 2;
    for (int i = threadIdx.x; i < OutH * InW * C; i += NTHREADS) {
        const int c = i % C;
        const int w = (i / C) % InW;
        const int h = i / (C * InW);
        const _Float16 x0 = in[((size_t)(2 * h) * InW + w) * C + c];
        const _Float16 x1 = in[((size_t)(2 * h + 1) * InW + w) * C + c];
        out[i] = ((float)x0 > (float)x1) ? x0 : x1;
    }
}

// ---------------- fused per-image CNN: one workgroup = one (b,t) image ----------------
__global__ __launch_bounds__(NTHREADS, 1) void fused_cnn(
    const float* __restrict__ x,       // (16,128,84) f32
    const _Float16* __restrict__ wpk,  // prepacked f16 weights (fragment order)
    const float* __restrict__ bias,    // folded biases
    float* __restrict__ outp)          // (2048, 26) f32
{
    __shared__ _Float16 s_win[84 * 15];           //  2.5 KB  (84x15x1)
    __shared__ _Float16 s_actA[84 * 15 * 32];     // 80.6 KB
    __shared__ _Float16 s_actB[84 * 15 * 32];     // 80.6 KB

    const int img = blockIdx.x;          // 0..2047
    const int bb  = img >> 7;            // batch
    const int tt  = img & 127;           // timestep

    // window extraction: img[h=f][w] = x[bb, tt + w - 7, f], zero-padded in time
    for (int i = threadIdx.x; i < 84 * 15; i += NTHREADS) {
        const int f = i / 15;
        const int w = i % 15;
        const int tsrc = tt + w - 7;
        float v = 0.0f;
        if (tsrc >= 0 && tsrc < 128) v = x[((size_t)bb * 128 + tsrc) * 84 + f];
        s_win[i] = (_Float16)v;
    }
    __syncthreads();

    // block1: 4x (conv3x3 pad1 + BN + ReLU), 84x15
    conv_wmma<84,15, 1, 84,15, 32, 3,3,1,true >(s_win,  s_actA, wpk + WOFF[0], bias + BOFF[0]); __syncthreads();
    conv_wmma<84,15,32, 84,15, 32, 3,3,1,true >(s_actA, s_actB, wpk + WOFF[1], bias + BOFF[1]); __syncthreads();
    conv_wmma<84,15,32, 84,15, 32, 3,3,1,true >(s_actB, s_actA, wpk + WOFF[2], bias + BOFF[2]); __syncthreads();
    conv_wmma<84,15,32, 84,15, 32, 3,3,1,true >(s_actA, s_actB, wpk + WOFF[3], bias + BOFF[3]); __syncthreads();
    maxpool2<84,15,32>(s_actB, s_actA); __syncthreads();          // -> 42x15x32

    // block2: conv3x3 valid
    conv_wmma<42,15,32, 40,13, 64, 3,3,0,true >(s_actA, s_actB, wpk + WOFF[4], bias + BOFF[4]); __syncthreads();
    conv_wmma<40,13,64, 38,11, 64, 3,3,0,true >(s_actB, s_actA, wpk + WOFF[5], bias + BOFF[5]); __syncthreads();
    maxpool2<38,11,64>(s_actA, s_actB); __syncthreads();          // -> 19x11x64

    // block3: conv12x9 valid -> 8x3x128
    conv_wmma<19,11,64,  8, 3,128,12,9,0,true >(s_actB, s_actA, wpk + WOFF[6], bias + BOFF[6]); __syncthreads();
    // out: 1x1 conv, 26 (padded 32) channels, no BN/ReLU
    conv_wmma< 8, 3,128, 8, 3, 32, 1,1,0,false>(s_actA, s_actB, wpk + WOFF[7], bias + BOFF[7]); __syncthreads();

    // spatial mean over 8x3 = 24 positions
    if (threadIdx.x < 26) {
        float s = 0.0f;
#pragma unroll
        for (int m = 0; m < 24; ++m) s += (float)s_actB[m * 32 + threadIdx.x];
        outp[(size_t)img * 26 + threadIdx.x] = s * (1.0f / 24.0f);
    }
}

// ---------------- host launch ----------------
extern "C" void kernel_launch(void* const* d_in, const int* in_sizes, int n_in,
                              void* d_out, int out_size, void* d_ws, size_t ws_size,
                              hipStream_t stream) {
    static const int hCin[8]   = {1, 32, 32, 32, 32, 64, 64, 128};
    static const int hKH[8]    = {3, 3, 3, 3, 3, 3, 12, 1};
    static const int hKW[8]    = {3, 3, 3, 3, 3, 3, 9, 1};
    static const int hCout[8]  = {32, 32, 32, 32, 64, 64, 128, 26};
    static const int hCoutP[8] = {32, 32, 32, 32, 64, 64, 128, 32};

    const float* x = (const float*)d_in[0];
    _Float16* wpk = (_Float16*)d_ws;
    float* bpk = (float*)((char*)d_ws + BIAS_BYTE);

    for (int l = 0; l < 8; ++l) {
        const float *w, *b, *g, *be, *me, *va;
        int has_bn;
        if (l < 7) {
            const int base = 1 + 6 * l;  // x, then 6 arrays per bn_conv
            w  = (const float*)d_in[base + 0];
            b  = (const float*)d_in[base + 1];
            g  = (const float*)d_in[base + 2];
            be = (const float*)d_in[base + 3];
            me = (const float*)d_in[base + 4];
            va = (const float*)d_in[base + 5];
            has_bn = 1;
        } else {
            w = (const float*)d_in[43];   // out_w (26,128,1,1)
            b = (const float*)d_in[44];   // out_b (26)
            g = be = me = va = b;         // unused (has_bn = 0)
            has_bn = 0;
        }
        const int K      = hCin[l] * hKH[l] * hKW[l];
        const int Ksteps = (K + 31) / 32;
        const int total  = (hCoutP[l] / 16) * Ksteps * 512;
        const int grid   = (total + NTHREADS - 1) / NTHREADS;
        prepack_layer<<<grid, NTHREADS, 0, stream>>>(
            w, b, g, be, me, va, hCin[l], hKH[l], hKW[l], hCout[l], hCoutP[l],
            wpk + WOFF[l], bpk + BOFF[l], has_bn);
    }

    fused_cnn<<<2048, NTHREADS, 0, stream>>>(x, wpk, bpk, (float*)d_out);
}
